// ProductOfGaussians_721554506119
// MI455X (gfx1250) — compile-verified, matrix operand order checked by reading, not run on hardware
//
#include <hip/hip_runtime.h>
#include <math.h>

#define T_LEN 8192
#define DIM_X 128
#define DIM_Z 16
#define DIM_H 512

typedef __attribute__((ext_vector_type(16))) __bf16 v16bf;
typedef __attribute__((ext_vector_type(8)))  __bf16 bf16x8;
typedef __attribute__((ext_vector_type(8)))  float  v8f;

// ---------------- conversion kernels ----------------
__global__ void cvt_f32_bf16(const float* __restrict__ in, __bf16* __restrict__ out, int n) {
  int i = blockIdx.x * blockDim.x + threadIdx.x;
  if (i < n) out[i] = (__bf16)in[i];
}

// W (K x N) f32 -> Wt (N x K) bf16
__global__ void transpose_cvt(const float* __restrict__ W, __bf16* __restrict__ Wt, int K, int N) {
  int idx = blockIdx.x * blockDim.x + threadIdx.x;
  if (idx >= K * N) return;
  int n = idx / K, k = idx - n * K;
  Wt[idx] = (__bf16)W[(size_t)k * N + n];
}

// ---------------- WMMA GEMM, 32x32 register-blocked ----------------
// out = act(A[M x K] @ Bt[N x K]^T + bias[N]); one wave per 32x32 macro-tile:
// 2 A fragments + 2 B fragments -> 4 WMMAs per K-step (2x arithmetic intensity
// vs 1 tile/wave).  A-fragment (16x32 bf16): lane L holds row L%16; its 16
// elements are two contiguous 8-elem runs at K = 8*(L/16) and 16 + 8*(L/16),
// so fragments load as two global_load_b128 from the row-major (pre-transposed
// for B) operand.
__device__ __forceinline__ v16bf ldfrag(const __bf16* p) {
  bf16x8 lo = *(const bf16x8*)p;
  bf16x8 hi = *(const bf16x8*)(p + 16);
  return __builtin_shufflevector(lo, hi, 0,1,2,3,4,5,6,7,8,9,10,11,12,13,14,15);
}

__global__ __launch_bounds__(32) void wmma_gemm(const __bf16* __restrict__ A,
                                                const __bf16* __restrict__ Bt,
                                                const float* __restrict__ bias,
                                                float* __restrict__ outF,
                                                __bf16* __restrict__ outBf,
                                                int N, int K, int relu_bf16) {
  const int lane = threadIdx.x;
  const int r = lane & 15, half = lane >> 4;
  const int n0 = blockIdx.x * 32;
  const int m0 = blockIdx.y * 32;
  const bool hasN1 = (n0 + 16) < N;
  const __bf16* a0p = A  + (size_t)(m0 + r) * K + half * 8;
  const __bf16* a1p = A  + (size_t)(m0 + 16 + r) * K + half * 8;
  const __bf16* b0p = Bt + (size_t)(n0 + r) * K + half * 8;
  const __bf16* b1p = Bt + (size_t)(n0 + (hasN1 ? 16 : 0) + r) * K + half * 8;
  v8f c00 = {}, c01 = {}, c10 = {}, c11 = {};
  for (int k0 = 0; k0 < K; k0 += 32) {
    if (k0 + 32 < K) {
      __builtin_prefetch(a0p + k0 + 32, 0, 0);   // -> global_prefetch_b8
      __builtin_prefetch(a1p + k0 + 32, 0, 0);
      __builtin_prefetch(b0p + k0 + 32, 0, 0);
      __builtin_prefetch(b1p + k0 + 32, 0, 0);
    }
    v16bf a0 = ldfrag(a0p + k0);
    v16bf a1 = ldfrag(a1p + k0);
    v16bf b0 = ldfrag(b0p + k0);
    c00 = __builtin_amdgcn_wmma_f32_16x16x32_bf16(false, a0, false, b0, (short)0, c00, false, false);
    c10 = __builtin_amdgcn_wmma_f32_16x16x32_bf16(false, a1, false, b0, (short)0, c10, false, false);
    if (hasN1) {
      v16bf b1 = ldfrag(b1p + k0);
      c01 = __builtin_amdgcn_wmma_f32_16x16x32_bf16(false, a0, false, b1, (short)0, c01, false, false);
      c11 = __builtin_amdgcn_wmma_f32_16x16x32_bf16(false, a1, false, b1, (short)0, c11, false, false);
    }
  }
  const int col0 = n0 + r;
  const float bn0 = bias[col0];
  const int col1 = n0 + 16 + r;
  const float bn1 = hasN1 ? bias[col1] : 0.0f;
  #pragma unroll
  for (int v = 0; v < 8; ++v) {
    int row0 = m0 + v + 8 * half;        // C/D layout: M = vgpr + 8*(lane/16)
    int row1 = row0 + 16;
    float v00 = c00[v] + bn0, v10 = c10[v] + bn0;
    float v01 = c01[v] + bn1, v11 = c11[v] + bn1;
    if (relu_bf16) {
      outBf[(size_t)row0 * N + col0] = (__bf16)fmaxf(v00, 0.0f);
      outBf[(size_t)row1 * N + col0] = (__bf16)fmaxf(v10, 0.0f);
      if (hasN1) {
        outBf[(size_t)row0 * N + col1] = (__bf16)fmaxf(v01, 0.0f);
        outBf[(size_t)row1 * N + col1] = (__bf16)fmaxf(v11, 0.0f);
      }
    } else {
      outF[(size_t)row0 * N + col0] = v00;
      outF[(size_t)row1 * N + col0] = v10;
      if (hasN1) {
        outF[(size_t)row0 * N + col1] = v01;
        outF[(size_t)row1 * N + col1] = v11;
      }
    }
  }
}

// ---------------- small constant matrices ----------------
// Qinv = Qc Qc^T, Q0inv = Q0c Q0c^T, M1 = A^T Qinv, AQA = M1 A
// C0 = Q0inv + AQA, Cmid = AQA + Qinv, Cend = Qinv, Bconst = -M1
__global__ __launch_bounds__(256) void build_consts(const float* __restrict__ A,
                                                    const float* __restrict__ Qc,
                                                    const float* __restrict__ Q0c,
                                                    float* __restrict__ C0,
                                                    float* __restrict__ Cmid,
                                                    float* __restrict__ Cend,
                                                    float* __restrict__ Bconst) {
  __shared__ float sA[16][16], sQc[16][16], sQ0c[16][16], sQinv[16][16], sM1[16][16];
  int tid = threadIdx.x, i = tid >> 4, j = tid & 15;
  sA[i][j] = A[tid]; sQc[i][j] = Qc[tid]; sQ0c[i][j] = Q0c[tid];
  __syncthreads();
  float qinv = 0.f, q0inv = 0.f;
  for (int k = 0; k < 16; ++k) { qinv += sQc[i][k] * sQc[j][k]; q0inv += sQ0c[i][k] * sQ0c[j][k]; }
  sQinv[i][j] = qinv;
  __syncthreads();
  float m1 = 0.f;
  for (int k = 0; k < 16; ++k) m1 += sA[k][i] * sQinv[k][j];
  sM1[i][j] = m1;
  __syncthreads();
  float aqa = 0.f;
  for (int k = 0; k < 16; ++k) aqa += sM1[i][k] * sA[k][j];
  C0[tid]     = q0inv + aqa;
  Cmid[tid]   = aqa + qinv;
  Cend[tid]   = qinv;
  Bconst[tid] = -m1;
}

// ---------------- per-timestep Lambda / AA / lambdaMu ----------------
__global__ __launch_bounds__(256) void build_AA(const float* __restrict__ Lc,
                                                const float* __restrict__ mean,
                                                const float* __restrict__ C0,
                                                const float* __restrict__ Cmid,
                                                const float* __restrict__ Cend,
                                                float* __restrict__ AA,
                                                float* __restrict__ lamMu, int T) {
  int t = blockIdx.x, tid = threadIdx.x, i = tid >> 4, j = tid & 15;
  __shared__ float sLc[16][16], sLam[16][16], sMean[16];
  sLc[i][j] = Lc[(size_t)t * 256 + tid];
  if (tid < 16) sMean[tid] = mean[(size_t)t * 16 + tid];
  __syncthreads();
  float lam = 0.f;
  #pragma unroll
  for (int k = 0; k < 16; ++k) lam += sLc[i][k] * sLc[j][k];
  sLam[i][j] = lam;
  const float* cb = (t == 0) ? C0 : ((t == T - 1) ? Cend : Cmid);
  AA[(size_t)t * 256 + tid] = lam + cb[tid];
  __syncthreads();
  if (j == 0) {
    float acc = 0.f;
    #pragma unroll
    for (int k = 0; k < 16; ++k) acc += sLam[i][k] * sMean[k];
    lamMu[(size_t)t * 16 + i] = acc;
  }
}

// ---------------- single-wave sequential solver ----------------
// 32 lanes; barriers in a single-wave workgroup lower to S_NOP (ISA 3.1),
// so the latency-bound 8192-step dependency chain pays no barrier cost.
__device__ inline void chol16(float S[16][16], int lane) {
  for (int k = 0; k < 16; ++k) {
    if (lane < 16 && lane >= k) {
      float d = S[k][k];                    // all lanes load before any store
      float s = sqrtf(d);
      S[lane][k] = (lane == k) ? s : S[lane][k] / s;
    }
    __syncthreads();
    for (int e = lane; e < 256; e += 32) {
      int i = e >> 4, j = e & 15;
      if (j > k && j <= i) S[i][j] -= S[i][k] * S[j][k];
    }
    __syncthreads();
  }
}

__device__ inline void trsm16(const float L[16][16], float Y[16][16], int lane) {
  for (int k = 0; k < 16; ++k) {
    if (lane < 16) Y[k][lane] /= L[k][k];
    __syncthreads();
    for (int e = lane; e < 256; e += 32) {
      int i = e >> 4, j = e & 15;
      if (i > k) Y[i][j] -= L[i][k] * Y[k][j];
    }
    __syncthreads();
  }
}

__device__ inline void fsolve16(const float L[16][16], float* rv, float* xv, int lane) {
  for (int k = 0; k < 16; ++k) {
    if (lane == k) xv[k] = rv[k] / L[k][k];
    __syncthreads();
    if (lane < 16 && lane > k) rv[lane] -= L[lane][k] * xv[k];
    __syncthreads();
  }
}

__device__ inline void bsolve16(const float L[16][16], float* rv, float* xv, int lane) {
  for (int k = 15; k >= 0; --k) {
    if (lane == k) xv[k] = rv[k] / L[k][k];
    __syncthreads();
    if (lane < k) rv[lane] -= L[k][lane] * xv[k];   // (L^T)[lane][k] = L[k][lane]
    __syncthreads();
  }
}

__global__ __launch_bounds__(32) void seq_solver(const float* __restrict__ AA,
                                                 const float* __restrict__ Bconst,
                                                 const float* __restrict__ lamMu,
                                                 float* __restrict__ Ld,
                                                 float* __restrict__ Cs,
                                                 float* __restrict__ ib,
                                                 float* __restrict__ out) {
  const int lane = threadIdx.x;
  __shared__ float Lp[16][16], Yc[16][16], Sm[16][16], Bc[16][16], Lb[16][16];
  __shared__ float xv[16], rv[16], red[16];
  float* LpF = &Lp[0][0]; float* YcF = &Yc[0][0]; float* SmF = &Sm[0][0];
  float* BcF = &Bc[0][0]; float* LbF = &Lb[0][0];

  for (int e = lane; e < 256; e += 32) { BcF[e] = Bconst[e]; SmF[e] = AA[e]; }
  __syncthreads();
  float logacc = 0.0f;

  // ---- phase 1: block-tridiagonal Cholesky chain ----
  chol16(Sm, lane);
  for (int e = lane; e < 256; e += 32) { Ld[e] = SmF[e]; LpF[e] = SmF[e]; }
  if (lane < 16) logacc += logf(Sm[lane][lane]);
  __syncthreads();

  for (int t = 1; t < T_LEN; ++t) {
    for (int e = lane; e < 256; e += 32) { YcF[e] = BcF[e]; SmF[e] = AA[(size_t)t * 256 + e]; }
    __syncthreads();
    trsm16(Lp, Yc, lane);                         // Y = Lprev^{-1} B;  Ci = Y^T
    for (int e = lane; e < 256; e += 32) {
      int i = e >> 4, j = e & 15;
      Cs[(size_t)(t - 1) * 256 + e] = Yc[j][i];   // Cs[i][j] = Ci[i][j]
      if (j <= i) {
        float acc = 0.0f;
        #pragma unroll
        for (int k = 0; k < 16; ++k) acc += Yc[k][i] * Yc[k][j];   // (Ci Ci^T)[i][j]
        Sm[i][j] -= acc;
      }
    }
    __syncthreads();
    chol16(Sm, lane);
    for (int e = lane; e < 256; e += 32) { Ld[(size_t)t * 256 + e] = SmF[e]; LpF[e] = SmF[e]; }
    if (lane < 16) logacc += logf(Sm[lane][lane]);
    __syncthreads();
  }

  // ---- phase 2: forward solve L x = lambdaMu ----
  if (lane < 16) rv[lane] = lamMu[lane];
  for (int e = lane; e < 256; e += 32) LbF[e] = Ld[e];
  __syncthreads();
  fsolve16(Lb, rv, xv, lane);
  if (lane < 16) ib[lane] = xv[lane];
  __syncthreads();
  for (int t = 1; t < T_LEN; ++t) {
    if (lane < 16) {
      float acc = 0.0f;
      #pragma unroll
      for (int k = 0; k < 16; ++k) acc += Cs[(size_t)(t - 1) * 256 + lane * 16 + k] * xv[k];
      rv[lane] = lamMu[(size_t)t * 16 + lane] - acc;
    }
    for (int e = lane; e < 256; e += 32) LbF[e] = Ld[(size_t)t * 256 + e];
    __syncthreads();
    fsolve16(Lb, rv, xv, lane);
    if (lane < 16) ib[(size_t)t * 16 + lane] = xv[lane];
    __syncthreads();
  }

  // ---- phase 3: backward solve L^T x = ib ----
  if (lane < 16) rv[lane] = ib[(size_t)(T_LEN - 1) * 16 + lane];
  for (int e = lane; e < 256; e += 32) LbF[e] = Ld[(size_t)(T_LEN - 1) * 256 + e];
  __syncthreads();
  bsolve16(Lb, rv, xv, lane);
  if (lane < 16) out[(size_t)(T_LEN - 1) * 16 + lane] = xv[lane];
  __syncthreads();
  for (int t = T_LEN - 2; t >= 0; --t) {
    if (lane < 16) {
      float acc = 0.0f;
      #pragma unroll
      for (int k = 0; k < 16; ++k) acc += Cs[(size_t)t * 256 + k * 16 + lane] * xv[k];  // Ci^T x
      rv[lane] = ib[(size_t)t * 16 + lane] - acc;
    }
    for (int e = lane; e < 256; e += 32) LbF[e] = Ld[(size_t)t * 256 + e];
    __syncthreads();
    bsolve16(Lb, rv, xv, lane);
    if (lane < 16) out[(size_t)t * 16 + lane] = xv[lane];
    __syncthreads();
  }

  // ---- ln_det = -2 * sum log diag ----
  if (lane < 16) red[lane] = logacc;
  __syncthreads();
  if (lane == 0) {
    float s = 0.0f;
    #pragma unroll
    for (int k = 0; k < 16; ++k) s += red[k];
    out[(size_t)T_LEN * 16] = -2.0f * s;
  }
}

// ---------------- host launch ----------------
extern "C" void kernel_launch(void* const* d_in, const int* in_sizes, int n_in,
                              void* d_out, int out_size, void* d_ws, size_t ws_size,
                              hipStream_t stream) {
  (void)in_sizes; (void)n_in; (void)out_size; (void)ws_size;
  const float* x      = (const float*)d_in[0];
  const float* Wm_in  = (const float*)d_in[1];
  const float* bm_in  = (const float*)d_in[2];
  const float* Wm_h1  = (const float*)d_in[3];
  const float* bm_h1  = (const float*)d_in[4];
  const float* Wm_h3  = (const float*)d_in[5];
  const float* bm_h3  = (const float*)d_in[6];
  const float* Wm_out = (const float*)d_in[7];
  const float* bm_out = (const float*)d_in[8];
  const float* Wc_in  = (const float*)d_in[9];
  const float* bc_in  = (const float*)d_in[10];
  const float* Wc_h1  = (const float*)d_in[11];
  const float* bc_h1  = (const float*)d_in[12];
  const float* Wc_h3  = (const float*)d_in[13];
  const float* bc_h3  = (const float*)d_in[14];
  const float* Wc_out = (const float*)d_in[15];
  const float* bc_out = (const float*)d_in[16];
  const float* A      = (const float*)d_in[17];
  const float* Qc     = (const float*)d_in[18];
  const float* Q0c    = (const float*)d_in[19];

  char* ws = (char*)d_ws;
  size_t off = 0;
  auto alloc = [&](size_t bytes) -> void* {
    void* p = ws + off;
    off += (bytes + 255) & ~(size_t)255;
    return p;
  };

  __bf16* xbf    = (__bf16*)alloc((size_t)T_LEN * DIM_X * 2);
  __bf16* WtMin  = (__bf16*)alloc((size_t)DIM_H * DIM_X * 2);
  __bf16* WtMh1  = (__bf16*)alloc((size_t)DIM_H * DIM_H * 2);
  __bf16* WtMh3  = (__bf16*)alloc((size_t)DIM_H * DIM_H * 2);
  __bf16* WtMout = (__bf16*)alloc((size_t)DIM_Z * DIM_H * 2);
  __bf16* WtCin  = (__bf16*)alloc((size_t)DIM_H * DIM_X * 2);
  __bf16* WtCh1  = (__bf16*)alloc((size_t)DIM_H * DIM_H * 2);
  __bf16* WtCh3  = (__bf16*)alloc((size_t)DIM_H * DIM_H * 2);
  __bf16* WtCout = (__bf16*)alloc((size_t)DIM_Z * DIM_Z * DIM_H * 2);
  __bf16* hA     = (__bf16*)alloc((size_t)T_LEN * DIM_H * 2);
  __bf16* hB     = (__bf16*)alloc((size_t)T_LEN * DIM_H * 2);
  float*  mean   = (float*)alloc((size_t)T_LEN * DIM_Z * 4);
  float*  Lc     = (float*)alloc((size_t)T_LEN * 256 * 4);
  float*  C0     = (float*)alloc(256 * 4);
  float*  Cmid   = (float*)alloc(256 * 4);
  float*  Cend   = (float*)alloc(256 * 4);
  float*  Bconst = (float*)alloc(256 * 4);
  float*  AAbuf  = (float*)alloc((size_t)T_LEN * 256 * 4);
  float*  lamMu  = (float*)alloc((size_t)T_LEN * DIM_Z * 4);
  float*  Ld     = (float*)alloc((size_t)T_LEN * 256 * 4);
  float*  Cs     = (float*)alloc((size_t)T_LEN * 256 * 4);
  float*  ibuf   = (float*)alloc((size_t)T_LEN * DIM_Z * 4);

  { int n = T_LEN * DIM_X;
    cvt_f32_bf16<<<(n + 255) / 256, 256, 0, stream>>>(x, xbf, n); }

  auto tr = [&](const float* W, __bf16* Wt, int K, int N) {
    int n = K * N;
    transpose_cvt<<<(n + 255) / 256, 256, 0, stream>>>(W, Wt, K, N);
  };
  tr(Wm_in, WtMin, DIM_X, DIM_H);
  tr(Wm_h1, WtMh1, DIM_H, DIM_H);
  tr(Wm_h3, WtMh3, DIM_H, DIM_H);
  tr(Wm_out, WtMout, DIM_H, DIM_Z);
  tr(Wc_in, WtCin, DIM_X, DIM_H);
  tr(Wc_h1, WtCh1, DIM_H, DIM_H);
  tr(Wc_h3, WtCh3, DIM_H, DIM_H);
  tr(Wc_out, WtCout, DIM_H, DIM_Z * DIM_Z);

  auto gemm = [&](const __bf16* Am, const __bf16* Bt, const float* bias,
                  float* outF, __bf16* outBf, int N, int K, int relu_bf) {
    dim3 grid((N + 31) / 32, T_LEN / 32);
    wmma_gemm<<<grid, 32, 0, stream>>>(Am, Bt, bias, outF, outBf, N, K, relu_bf);
  };
  // mean encoder
  gemm(xbf, WtMin, bm_in, nullptr, hA, DIM_H, DIM_X, 1);
  gemm(hA, WtMh1, bm_h1, nullptr, hB, DIM_H, DIM_H, 1);
  gemm(hB, WtMh3, bm_h3, nullptr, hA, DIM_H, DIM_H, 1);
  gemm(hA, WtMout, bm_out, mean, nullptr, DIM_Z, DIM_H, 0);
  // cov encoder
  gemm(xbf, WtCin, bc_in, nullptr, hA, DIM_H, DIM_X, 1);
  gemm(hA, WtCh1, bc_h1, nullptr, hB, DIM_H, DIM_H, 1);
  gemm(hB, WtCh3, bc_h3, nullptr, hA, DIM_H, DIM_H, 1);
  gemm(hA, WtCout, bc_out, Lc, nullptr, DIM_Z * DIM_Z, DIM_H, 0);

  build_consts<<<1, 256, 0, stream>>>(A, Qc, Q0c, C0, Cmid, Cend, Bconst);
  build_AA<<<T_LEN, 256, 0, stream>>>(Lc, mean, C0, Cmid, Cend, AAbuf, lamMu, T_LEN);
  seq_solver<<<1, 32, 0, stream>>>(AAbuf, Bconst, lamMu, Ld, Cs, ibuf, (float*)d_out);
}
